// RotarySelfAttention_60790967108349
// MI455X (gfx1250) — matrix operand order repaired and checked by reading, not compiled, not run on hardware
//
#include <hip/hip_runtime.h>
#include <math.h>

// ---------------------------------------------------------------------------
// Problem constants (from reference): B=2, S=2048, D=512, 8 heads of 64.
// ---------------------------------------------------------------------------
#define S_LEN   2048
#define D_MODEL 512
#define NHEAD   8
#define HDIM    64
#define BATCH   2
#define ROWS    (BATCH * S_LEN)      // 4096
#define QKV_N   (3 * D_MODEL)        // 1536

typedef __attribute__((ext_vector_type(16))) __bf16 v16bf;
typedef __attribute__((ext_vector_type(8)))  float  v8f;

// ---------------------------------------------------------------------------
// Helpers
// ---------------------------------------------------------------------------
__device__ __forceinline__ unsigned short f2bf(float f) {
    unsigned int u = __float_as_uint(f);
    u += 0x7FFFu + ((u >> 16) & 1u);           // round-to-nearest-even
    return (unsigned short)(u >> 16);
}

// Assemble a 16-element bf16 WMMA fragment from two 16-byte chunks.
__device__ __forceinline__ v16bf load_frag(const unsigned short* p0,
                                           const unsigned short* p1) {
    union { uint4 u[2]; v16bf v; } f;
    f.u[0] = *reinterpret_cast<const uint4*>(p0);
    f.u[1] = *reinterpret_cast<const uint4*>(p1);
    return f.v;
}

__device__ __forceinline__ v8f wmma_bf16(v16bf a, v16bf b, v8f c) {
    // D = A(16x32 bf16) * B(32x16 bf16) + C(16x16 f32)
    return __builtin_amdgcn_wmma_f32_16x16x32_bf16(
        /*neg_a=*/false, a, /*neg_b=*/false, b,
        /*c_mod=*/(short)0, c, /*reuse_a=*/false, /*reuse_b=*/false);
}

// ---------------------------------------------------------------------------
// Kernel 1a: f32 -> bf16 elementwise (for x)
// ---------------------------------------------------------------------------
__global__ void k_convert_f32_bf16(const float* __restrict__ in,
                                   unsigned short* __restrict__ out, int n) {
    int i = blockIdx.x * 256 + threadIdx.x;
    if (i < n) out[i] = f2bf(in[i]);
}

// ---------------------------------------------------------------------------
// Kernel 1b: transpose weight (K x N, f32, row-major) -> (N x K, bf16)
// so that WMMA B-fragments become K-contiguous b128 loads.
// ---------------------------------------------------------------------------
__global__ void k_transpose_w(const float* __restrict__ w,
                              unsigned short* __restrict__ wt, int K, int N) {
    int i = blockIdx.x * 256 + threadIdx.x;
    if (i < K * N) {
        int n = i / K;
        int k = i - n * K;
        wt[(size_t)n * K + k] = f2bf(w[(size_t)k * N + n]);
    }
}

// ---------------------------------------------------------------------------
// Kernel 2: qkv = x @ Wqkv + bqkv, then RoPE on q/k, scatter to head layout.
//   x_bf : (4096 x 512) bf16 row-major
//   wt   : (1536 x 512) bf16 (transposed weight, K contiguous)
//   qb,kb: [b*8+h][s][64] bf16
//   vt   : [b*8+h][64][s] bf16 (transposed for attn@V B-fragments)
// One wave = 16 rows x 64 cols (4 N-tiles, A-fragment reused 4x).
// ---------------------------------------------------------------------------
__global__ __launch_bounds__(128) void
k_qkv_rope(const unsigned short* __restrict__ xb,
           const unsigned short* __restrict__ wt,
           const float* __restrict__ bias,
           unsigned short* __restrict__ qb,
           unsigned short* __restrict__ kb,
           unsigned short* __restrict__ vt) {
    const int lane = threadIdx.x & 31;
    const int wave = threadIdx.x >> 5;
    const int lr   = lane & 15;
    const int half = lane >> 4;

    const int n0   = (blockIdx.x * 4 + wave) * 64;   // 64-wide output span
    const int row0 = blockIdx.y * 16;                // row (b*S + s) tile

    v8f acc[4];
    {
        v8f z = {};
#pragma unroll
        for (int t = 0; t < 4; ++t) acc[t] = z;
    }

    // A-fragment: lane<16 holds K {0..7,16..23}; lane>=16 holds {8..15,24..31}
    const unsigned short* arow  = xb + (size_t)(row0 + lr) * D_MODEL + (half ? 8 : 0);
    // B-fragment: lane<16 holds K {0..7,16..23}; lane>=16 holds {8..15,24..31}
    const unsigned short* bbase = wt + (size_t)(n0 + lr) * D_MODEL + (half ? 8 : 0);

    for (int kk = 0; kk < D_MODEL; kk += 32) {
        v16bf a = load_frag(arow + kk, arow + kk + 16);
#pragma unroll
        for (int t = 0; t < 4; ++t) {
            const unsigned short* bp = bbase + (size_t)t * 16 * D_MODEL + kk;
            v16bf b = load_frag(bp, bp + 16);
            acc[t] = wmma_bf16(a, b, acc[t]);
        }
    }

#pragma unroll
    for (int t = 0; t < 4; ++t) {
        const int n   = n0 + 16 * t + lr;   // global output feature 0..1535
        const float bv = bias[n];
        const int sec = n / D_MODEL;        // 0=q 1=k 2=v (uniform per wave,t)
        const int sc  = n % D_MODEL;
        const int h   = sc / HDIM;
        const int d   = sc % HDIM;

        // RoPE: theta = s * 10000^(-(2i)/D), 2i = sc & ~1 (same for lane pair)
        const float freq = __expf(-(float)(sc & ~1) * (9.2103403719761836f / 512.0f));
        const float sgn  = (n & 1) ? 1.0f : -1.0f;

#pragma unroll
        for (int r = 0; r < 8; ++r) {
            const int row = row0 + r + 8 * half;   // global row in [0,4096)
            const int b_i = row >> 11;             // batch
            const int s   = row & (S_LEN - 1);     // sequence pos
            float val = acc[t][r] + bv;
            if (sec < 2) {
                float sn, cs;
                __sincosf((float)s * freq, &sn, &cs);
                float part = __shfl_xor(val, 1, 32);   // rope pair partner
                val = val * cs + sgn * part * sn;      // even: x1c-x2s ; odd: x1s+x2c
                unsigned short* dst = sec ? kb : qb;
                dst[((size_t)(b_i * NHEAD + h) * S_LEN + s) * HDIM + d] = f2bf(val);
            } else {
                vt[((size_t)(b_i * NHEAD + h) * HDIM + d) * S_LEN + s] = f2bf(val);
            }
        }
    }
}

// ---------------------------------------------------------------------------
// Kernel 3: flash-style attention per (b,h).
// Block = 4 waves sharing (b,h); K/V tiles staged once per block in LDS.
// Each wave owns a 16-query tile, streams keys in blocks of 32 with online
// softmax; probabilities re-striped C-layout -> A-layout via private LDS.
// ---------------------------------------------------------------------------
__global__ __launch_bounds__(128) void
k_attention(const unsigned short* __restrict__ qb,
            const unsigned short* __restrict__ kb,
            const unsigned short* __restrict__ vt,
            unsigned short* __restrict__ ob) {
    __shared__ __align__(16) unsigned short Kt[32 * HDIM];     // 4 KB: [key][d]
    __shared__ __align__(16) unsigned short Vt[HDIM * 32];     // 4 KB: [d][key]
    __shared__ __align__(16) unsigned short Plds[4][16 * 32];  // 1 KB / wave

    const int tid  = threadIdx.x;
    const int lane = tid & 31;
    const int wave = tid >> 5;
    const int lr   = lane & 15;
    const int half = lane >> 4;

    const int s0 = (blockIdx.x * 4 + wave) * 16;  // query tile start
    const int bh = blockIdx.y;                    // b*8 + h

    // Q A-fragments for both 32-wide K(=head_dim) chunks, loaded once.
    const unsigned short* qrow = qb + ((size_t)bh * S_LEN + s0 + lr) * HDIM;
    const int asel = half ? 8 : 0;
    v16bf qa0 = load_frag(qrow + asel,      qrow + asel + 16);
    v16bf qa1 = load_frag(qrow + 32 + asel, qrow + 32 + asel + 16);

    float m[8], l[8];
    v8f acc[4];
    {
        v8f z = {};
#pragma unroll
        for (int t = 0; t < 4; ++t) acc[t] = z;
#pragma unroll
        for (int r = 0; r < 8; ++r) { m[r] = -1.0e30f; l[r] = 0.0f; }
    }

    unsigned short* myP = Plds[wave];
    const int bsel = half ? 8 : 0;

    for (int kk = 0; kk < S_LEN; kk += 32) {
        __syncthreads();  // previous iteration's LDS tile reads complete

        // ---- cooperative stage of K (32x64) and V^T (64x32) tiles --------
        // Both tiles are 256 uint4; 128 threads copy 2 uint4 each per tile.
#pragma unroll
        for (int j = 0; j < 2; ++j) {
            const int idx = tid + 128 * j;
            reinterpret_cast<uint4*>(Kt)[idx] = *reinterpret_cast<const uint4*>(
                kb + ((size_t)bh * S_LEN + kk + (idx >> 3)) * HDIM + (idx & 7) * 8);
            reinterpret_cast<uint4*>(Vt)[idx] = *reinterpret_cast<const uint4*>(
                vt + ((size_t)bh * HDIM + (idx >> 2)) * S_LEN + kk + (idx & 3) * 8);
        }
        if (kk + 32 < S_LEN) {
            __builtin_prefetch(kb + ((size_t)bh * S_LEN + kk + 32 + lr) * HDIM, 0, 0);
            __builtin_prefetch(vt + ((size_t)bh * HDIM + lane) * S_LEN + kk + 32, 0, 0);
        }
        __syncthreads();

        // ---- scores: 16 queries x 32 keys, two 16x16 tiles, K-dim = 64 ----
        v8f st[2];
#pragma unroll
        for (int t = 0; t < 2; ++t) {
            const unsigned short* kr = Kt + (16 * t + lr) * HDIM + bsel;
            v16bf b0 = load_frag(kr,      kr + 16);   // d = 0..31 chunk
            v16bf b1 = load_frag(kr + 32, kr + 48);   // d = 32..63 chunk
            v8f z = {};
            z = wmma_bf16(qa0, b0, z);
            z = wmma_bf16(qa1, b1, z);
            st[t] = z;
        }
        // scale by 1/sqrt(64)
#pragma unroll
        for (int t = 0; t < 2; ++t)
#pragma unroll
            for (int r = 0; r < 8; ++r) st[t][r] *= 0.125f;

        // ---- online softmax: row reductions across the 16-lane halves ----
        float mn[8], alpha[8];
#pragma unroll
        for (int r = 0; r < 8; ++r) {
            float mr = fmaxf(st[0][r], st[1][r]);
            mr = fmaxf(mr, __shfl_xor(mr, 1, 32));
            mr = fmaxf(mr, __shfl_xor(mr, 2, 32));
            mr = fmaxf(mr, __shfl_xor(mr, 4, 32));
            mr = fmaxf(mr, __shfl_xor(mr, 8, 32));
            mn[r]    = fmaxf(m[r], mr);
            alpha[r] = __expf(m[r] - mn[r]);
            m[r]     = mn[r];
        }
#pragma unroll
        for (int t = 0; t < 2; ++t)
#pragma unroll
            for (int r = 0; r < 8; ++r) st[t][r] = __expf(st[t][r] - mn[r]);
#pragma unroll
        for (int r = 0; r < 8; ++r) {
            float rs = st[0][r] + st[1][r];
            rs += __shfl_xor(rs, 1, 32);
            rs += __shfl_xor(rs, 2, 32);
            rs += __shfl_xor(rs, 4, 32);
            rs += __shfl_xor(rs, 8, 32);
            l[r] = l[r] * alpha[r] + rs;
        }
#pragma unroll
        for (int t = 0; t < 4; ++t)
#pragma unroll
            for (int r = 0; r < 8; ++r) acc[t][r] *= alpha[r];

        // ---- restripe P: C-layout (lane=col) -> A-layout (lane=row) via LDS
#pragma unroll
        for (int t = 0; t < 2; ++t)
#pragma unroll
            for (int r = 0; r < 8; ++r)
                myP[(r + 8 * half) * 32 + lr + 16 * t] = f2bf(st[t][r]);

        asm volatile("s_wait_dscnt 0" ::: "memory");  // same-wave LDS RAW fence

        v16bf pa = half ? load_frag(&myP[lr * 32 + 8], &myP[lr * 32 + 24])
                        : load_frag(&myP[lr * 32 + 0], &myP[lr * 32 + 16]);

        // ---- out += P (16x32) @ V (32x64): 4 WMMAs over head-dim tiles ----
#pragma unroll
        for (int t = 0; t < 4; ++t) {
            const unsigned short* vr = Vt + (t * 16 + lr) * 32 + bsel;
            v16bf vb = load_frag(vr, vr + 16);
            acc[t] = wmma_bf16(pa, vb, acc[t]);
        }
    }

    // ---- epilogue: normalize by row sum, write (B,S,D) bf16 ----
    const int b_i = bh >> 3, h = bh & 7;
#pragma unroll
    for (int r = 0; r < 8; ++r) {
        const float inv = 1.0f / l[r];
        const int row = s0 + r + 8 * half;
#pragma unroll
        for (int t = 0; t < 4; ++t) {
            ob[((size_t)(b_i * S_LEN + row)) * D_MODEL + h * HDIM + t * 16 + lr] =
                f2bf(acc[t][r] * inv);
        }
    }
}

// ---------------------------------------------------------------------------
// Kernel 4: final projection  out = attn_out @ Wo + bo  (f32 output)
// One wave = 16 rows x 64 cols (4 N-tiles, A-fragment reused 4x).
// ---------------------------------------------------------------------------
__global__ __launch_bounds__(128) void
k_out_proj(const unsigned short* __restrict__ ab,
           const unsigned short* __restrict__ wt,
           const float* __restrict__ bias,
           float* __restrict__ out) {
    const int lane = threadIdx.x & 31;
    const int wave = threadIdx.x >> 5;
    const int lr   = lane & 15;
    const int half = lane >> 4;

    const int n0   = (blockIdx.x * 4 + wave) * 64;
    const int row0 = blockIdx.y * 16;

    v8f acc[4];
    {
        v8f z = {};
#pragma unroll
        for (int t = 0; t < 4; ++t) acc[t] = z;
    }

    const unsigned short* arow  = ab + (size_t)(row0 + lr) * D_MODEL + (half ? 8 : 0);
    const unsigned short* bbase = wt + (size_t)(n0 + lr) * D_MODEL + (half ? 8 : 0);

    for (int kk = 0; kk < D_MODEL; kk += 32) {
        v16bf a = load_frag(arow + kk, arow + kk + 16);
#pragma unroll
        for (int t = 0; t < 4; ++t) {
            const unsigned short* bp = bbase + (size_t)t * 16 * D_MODEL + kk;
            v16bf b = load_frag(bp, bp + 16);
            acc[t] = wmma_bf16(a, b, acc[t]);
        }
    }

#pragma unroll
    for (int t = 0; t < 4; ++t) {
        const float bv = bias[n0 + 16 * t + lr];
#pragma unroll
        for (int r = 0; r < 8; ++r) {
            const int row = row0 + r + 8 * half;
            out[(size_t)row * D_MODEL + n0 + 16 * t + lr] = acc[t][r] + bv;
        }
    }
}

// ---------------------------------------------------------------------------
// Host-side launcher
// ---------------------------------------------------------------------------
extern "C" void kernel_launch(void* const* d_in, const int* in_sizes, int n_in,
                              void* d_out, int out_size, void* d_ws, size_t ws_size,
                              hipStream_t stream) {
    const float* x    = (const float*)d_in[0];   // (2,2048,512)
    const float* Wqkv = (const float*)d_in[1];   // (512,1536)
    const float* bqkv = (const float*)d_in[2];   // (1536,)
    const float* Wo   = (const float*)d_in[3];   // (512,512)
    const float* bo   = (const float*)d_in[4];   // (512,)
    float* out        = (float*)d_out;           // (2,2048,512) f32

    // workspace layout (bf16 buffers, all 16B aligned)
    char* w = (char*)d_ws;
    unsigned short* x_bf   = (unsigned short*)(w);                          // 4 MB
    unsigned short* wt_qkv = (unsigned short*)(w + 4194304);                // 1.5 MB
    unsigned short* wt_o   = (unsigned short*)(w + 4194304 + 1572864);      // 0.5 MB
    unsigned short* q_buf  = (unsigned short*)(w + 6291456);                // 4 MB
    unsigned short* k_buf  = (unsigned short*)(w + 10485760);               // 4 MB
    unsigned short* v_t    = (unsigned short*)(w + 14680064);               // 4 MB
    unsigned short* a_out  = (unsigned short*)(w + 18874368);               // 4 MB
    (void)in_sizes; (void)n_in; (void)out_size; (void)ws_size;

    // 1) precision conversion + weight transposes
    {
        int n = ROWS * D_MODEL;                               // 2,097,152
        k_convert_f32_bf16<<<(n + 255) / 256, 256, 0, stream>>>(x, x_bf, n);
    }
    {
        int n = D_MODEL * QKV_N;                              // 786,432
        k_transpose_w<<<(n + 255) / 256, 256, 0, stream>>>(Wqkv, wt_qkv, D_MODEL, QKV_N);
    }
    {
        int n = D_MODEL * D_MODEL;                            // 262,144
        k_transpose_w<<<(n + 255) / 256, 256, 0, stream>>>(Wo, wt_o, D_MODEL, D_MODEL);
    }

    // 2) QKV GEMM + bias + RoPE + head scatter  (grid: 1536/(4*64)=6 x 256)
    k_qkv_rope<<<dim3(QKV_N / 256, ROWS / 16), 128, 0, stream>>>(
        x_bf, wt_qkv, bqkv, q_buf, k_buf, v_t);

    // 3) attention (flash-style, online softmax, LDS-staged K/V)
    k_attention<<<dim3(S_LEN / 64, BATCH * NHEAD), 128, 0, stream>>>(
        q_buf, k_buf, v_t, a_out);

    // 4) output projection + bias  (grid: 512/(4*64)=2 x 256)
    k_out_proj<<<dim3(D_MODEL / 256, ROWS / 16), 128, 0, stream>>>(
        a_out, wt_o, bo, out);
}
